// LongAttention_89335319757204
// MI455X (gfx1250) — matrix operand order repaired and checked by reading, not compiled, not run on hardware
//
#include <hip/hip_runtime.h>
#include <hip/hip_bf16.h>

// ---------------- problem constants ----------------
#define BB 2
#define TT 4096
#define CC 2048
#define HH 16
#define DD 128            // CC / HH
#define KCONV 4
#define MM (BB * TT)      // 8192 rows in all GEMMs

typedef __attribute__((ext_vector_type(16))) __bf16 v16bf;
typedef __attribute__((ext_vector_type(8)))  __bf16 v8bf;
typedef __attribute__((ext_vector_type(8)))  float  v8f;
typedef __attribute__((ext_vector_type(4)))  unsigned int v4u;
typedef __attribute__((ext_vector_type(8)))  int v8i;
typedef __attribute__((ext_vector_type(4)))  int v4i;

__device__ __forceinline__ __bf16 f2bf(float f) {
    __hip_bfloat16 h = __float2bfloat16(f);
    __bf16 r;
    __builtin_memcpy(&r, &h, sizeof(r));
    return r;
}

// ---------------- TDM: 2D bf16 tile -> LDS (Tensor Data Mover) ----------------
// D# per CDNA5 ISA ch.8: group0 = {ctrl, lds_addr, gaddr_lo, gaddr_hi|type},
// group1 = dims/strides. Padding: pad_interval=64B (code 3), pad_amount=16B
// (code 3) -> LDS row pitch 80B (16B aligned, bank-conflict-free fragments).
__device__ __forceinline__ void tdm_issue(v4u g0, v8i g1) {
    v4i z4 = {0, 0, 0, 0};
#if defined(__clang_major__) && (__clang_major__ >= 23)
    v8i z8 = {0, 0, 0, 0, 0, 0, 0, 0};
    __builtin_amdgcn_tensor_load_to_lds(g0, g1, z4, z4, z8, 0);
#else
    __builtin_amdgcn_tensor_load_to_lds(g0, g1, z4, z4, 0);
#endif
}

__device__ __forceinline__ void tdm_load_tile_bf16(
    unsigned lds_off, const __bf16* gptr,
    unsigned tensor_d0, unsigned tensor_d1,
    unsigned tile_d0, unsigned tile_d1, unsigned row_stride_elems) {
    unsigned long long ga = (unsigned long long)(size_t)gptr;
    v4u g0;
    g0.x = 1u;                                            // count=1, user mode
    g0.y = lds_off;                                       // LDS byte address
    g0.z = (unsigned)(ga & 0xFFFFFFFFu);                  // global_addr[31:0]
    g0.w = (unsigned)((ga >> 32) & 0x1FFFFFFu) | (2u << 30); // [56:32] | type=2
    v8i g1;
    g1[0] = (int)((1u << 16)      // data_size = 2 bytes
                | (1u << 20)      // pad_enable
                | (3u << 22)      // pad_interval: 16 dwords (64B) per row
                | (3u << 25));    // pad_amount: 4 dwords (16B)
    g1[1] = (int)((tensor_d0 & 0xFFFFu) << 16);           // tensor_dim0[15:0]
    g1[2] = (int)((tensor_d0 >> 16) | ((tensor_d1 & 0xFFFFu) << 16));
    g1[3] = (int)((tensor_d1 >> 16) | (tile_d0 << 16));   // tile_dim0
    g1[4] = (int)(tile_d1 & 0xFFFFu);                     // tile_dim1, tile_dim2=0
    g1[5] = (int)row_stride_elems;                        // tensor_dim0_stride[31:0]
    g1[6] = 0;                                            // stride[47:32], dim1_stride lo
    g1[7] = 0;
    tdm_issue(g0, g1);
}

// ---------------- fp32 -> bf16 convert (weights) ----------------
__global__ void cvt_bf16_kernel(const float* __restrict__ in,
                                __bf16* __restrict__ out, long n) {
    long i = (long)blockIdx.x * blockDim.x + threadIdx.x;
    long stride = (long)gridDim.x * blockDim.x;
    for (; i < n; i += stride) out[i] = f2bf(in[i]);
}

// ---------------- causal depthwise conv + x->bf16 ----------------
__global__ void conv_kernel(const float* __restrict__ x,
                            const float* __restrict__ cw,   // [C,1,KCONV]
                            const float* __restrict__ cb,   // [C]
                            __bf16* __restrict__ xbf,
                            __bf16* __restrict__ xcbf) {
    long idx = (long)blockIdx.x * blockDim.x + threadIdx.x;
    const long total = (long)BB * TT * CC;
    if (idx >= total) return;
    int c = (int)(idx % CC);
    int t = (int)((idx / CC) % TT);
    long b = idx / ((long)TT * CC);
    float acc = cb[c];
#pragma unroll
    for (int j = 0; j < KCONV; ++j) {
        int tt = t - (KCONV - 1) + j;
        if (tt >= 0) acc += x[(b * TT + tt) * (long)CC + c] * cw[c * KCONV + j];
    }
    xcbf[idx] = f2bf(acc);
    xbf[idx]  = f2bf(x[idx]);
}

// ---------------- bf16 WMMA GEMM with TDM-staged LDS tiles ----------------
// Cout[M,N] = A[M,K] * Bw[N,K]^T.  mode 0: plain store. mode 1: sigmoid(+bias).
// Block: 256 threads (8 waves), tile 128x128, wave tile 32x64.
// Wave 0 DMAs the 128x32 A and B tiles into double-buffered LDS via
// tensor_load_to_lds; all waves read fragments with ds_load_b128.
#define LDS_PITCH 80                      // 64B row + 16B TDM pad
#define LDS_TILE  (128 * LDS_PITCH)      // 10240B per tile
__global__ __launch_bounds__(256) void gemm_bf16_nt(
    const __bf16* __restrict__ A,
    const __bf16* __restrict__ Bw,
    float* __restrict__ Cout,
    const float* __restrict__ bias,
    int M, int N, int Kd, int mode) {
    __shared__ __attribute__((aligned(16))) char smem[4 * LDS_TILE];  // A0 A1 B0 B1

    const int lane  = threadIdx.x & 31;
    const int wid   = threadIdx.x >> 5;      // 0..7
    const int waveM = wid >> 1;              // 0..3
    const int waveN = wid & 1;               // 0..1
    const int half  = lane >> 4;             // 0/1
    const int l15   = lane & 15;
    const int koffA = half * 8;              // A: elems {koffA..+7, koffA+16..+23}
    const int koffB = half * 8;              // B: elems {koffB..+7, koffB+16..+23}

    const int blockM = blockIdx.y * 128;
    const int blockN = blockIdx.x * 128;
    const unsigned smem_base = (unsigned)(size_t)&smem[0];  // LDS byte offset
    const int NS = Kd / 32;

    v8f acc[2][4];
#pragma unroll
    for (int mt = 0; mt < 2; ++mt)
#pragma unroll
        for (int nt = 0; nt < 4; ++nt) acc[mt][nt] = (v8f)(0.0f);

    // prologue: slab 0 -> buffer 0
    if (wid == 0) {
        const __bf16* gA = A  + (size_t)blockM * Kd;
        const __bf16* gB = Bw + (size_t)blockN * Kd;
        tdm_load_tile_bf16(smem_base,                gA, Kd, M, 32, 128, Kd);
        tdm_load_tile_bf16(smem_base + 2 * LDS_TILE, gB, Kd, N, 32, 128, Kd);
    }

    for (int ks = 0; ks < NS; ++ks) {
        if (wid == 0) {
            if (ks + 1 < NS) {
                const int kk = (ks + 1) * 32;
                const int nb = (ks + 1) & 1;
                const __bf16* gA = A  + (size_t)blockM * Kd + kk;
                const __bf16* gB = Bw + (size_t)blockN * Kd + kk;
                tdm_load_tile_bf16(smem_base + nb * LDS_TILE,       gA, Kd, M, 32, 128, Kd);
                tdm_load_tile_bf16(smem_base + (2 + nb) * LDS_TILE, gB, Kd, N, 32, 128, Kd);
                __builtin_amdgcn_s_wait_tensorcnt((short)2);  // slab ks landed
            } else {
                __builtin_amdgcn_s_wait_tensorcnt((short)0);
            }
        }
        __syncthreads();   // publish TDM-filled tile to all waves

        const char* aT = &smem[(ks & 1) * LDS_TILE];
        const char* bT = &smem[(2 + (ks & 1)) * LDS_TILE];

        v16bf afrag[2], bfrag[4];
#pragma unroll
        for (int mt = 0; mt < 2; ++mt) {
            const char* p = aT + (waveM * 32 + mt * 16 + l15) * LDS_PITCH + koffA * 2;
            v8bf lo = *(const v8bf*)(p);
            v8bf hi = *(const v8bf*)(p + 32);
            afrag[mt] = __builtin_shufflevector(lo, hi, 0,1,2,3,4,5,6,7,8,9,10,11,12,13,14,15);
        }
#pragma unroll
        for (int nt = 0; nt < 4; ++nt) {
            const char* p = bT + (waveN * 64 + nt * 16 + l15) * LDS_PITCH + koffB * 2;
            v8bf lo = *(const v8bf*)(p);
            v8bf hi = *(const v8bf*)(p + 32);
            bfrag[nt] = __builtin_shufflevector(lo, hi, 0,1,2,3,4,5,6,7,8,9,10,11,12,13,14,15);
        }
#pragma unroll
        for (int mt = 0; mt < 2; ++mt)
#pragma unroll
            for (int nt = 0; nt < 4; ++nt)
                acc[mt][nt] = __builtin_amdgcn_wmma_f32_16x16x32_bf16(
                    false, afrag[mt], false, bfrag[nt], (short)0, acc[mt][nt], false, false);

        __syncthreads();   // all reads done before wave0 overwrites this buffer
    }

    // epilogue: C/D layout -> lane<16: (m=r, n=l15); lane>=16: (m=8+r, n=l15)
#pragma unroll
    for (int mt = 0; mt < 2; ++mt) {
#pragma unroll
        for (int nt = 0; nt < 4; ++nt) {
            const int n = blockN + waveN * 64 + nt * 16 + l15;
            float bn = (mode == 1) ? bias[n] : 0.0f;
#pragma unroll
            for (int r = 0; r < 8; ++r) {
                const int m = blockM + waveM * 32 + mt * 16 + half * 8 + r;
                float v = acc[mt][nt][r];
                if (mode == 1) v = __builtin_amdgcn_rcpf(1.0f + __expf(-(v + bn)));
                Cout[(long)m * N + n] = v;
            }
        }
    }
}

// ---------------- l2norm(q), l2norm(k), kv = gate * kn * v ----------------
__global__ __launch_bounds__(256) void qknorm_kv_kernel(
    float* __restrict__ q, const float* __restrict__ k,
    float* __restrict__ v, const float* __restrict__ gate) {
    const int lane = threadIdx.x & 31;
    const long row = (long)blockIdx.x * 8 + (threadIdx.x >> 5);
    const long base = row * DD + lane * 4;
    float4 q4 = *(const float4*)(q + base);
    float4 k4 = *(const float4*)(k + base);
    float4 v4 = *(const float4*)(v + base);
    float4 g4 = *(const float4*)(gate + base);
    float sq = q4.x * q4.x + q4.y * q4.y + q4.z * q4.z + q4.w * q4.w;
    float sk = k4.x * k4.x + k4.y * k4.y + k4.z * k4.z + k4.w * k4.w;
#pragma unroll
    for (int off = 16; off > 0; off >>= 1) {
        sq += __shfl_xor(sq, off, 32);
        sk += __shfl_xor(sk, off, 32);
    }
    const float qi = 1.0f / fmaxf(sqrtf(sq), 1e-12f);
    const float ki = 1.0f / fmaxf(sqrtf(sk), 1e-12f);
    float4 qn = make_float4(q4.x * qi, q4.y * qi, q4.z * qi, q4.w * qi);
    float4 kv = make_float4(g4.x * k4.x * ki * v4.x, g4.y * k4.y * ki * v4.y,
                            g4.z * k4.z * ki * v4.z, g4.w * k4.w * ki * v4.w);
    *(float4*)(q + base) = qn;
    *(float4*)(v + base) = kv;
}

// ---------------- EMA scan over T, fused with u = mem*q -> bf16 ----------------
__global__ __launch_bounds__(256) void scan_kernel(
    const float* __restrict__ kv, const float* __restrict__ q,
    __bf16* __restrict__ ubf, float* __restrict__ state_out) {
    const int tid = blockIdx.x * blockDim.x + threadIdx.x;  // 0..B*C-1
    const int b = tid >> 11;
    const int cc = tid & 2047;
    const int h = cc >> 7;
    const float gam = (h == 0) ? 1.0f : (0.9f + (0.099f / 15.0f) * (float)h);
    const long base = (long)b * TT * CC + cc;
    float mem = 0.0f;
    for (int t = 0; t < TT; ++t) {
        const long o = base + (long)t * CC;
        mem = gam * mem + kv[o];
        ubf[o] = f2bf(mem * q[o]);
    }
    state_out[(long)b * CC + cc] = mem;
}

// ---------------- LayerNorm over C, one block per row ----------------
__global__ __launch_bounds__(256) void layernorm_kernel(
    const float* __restrict__ in, const float* __restrict__ w,
    const float* __restrict__ bb, float* __restrict__ out) {
    const long row = blockIdx.x;
    const int tid = threadIdx.x;
    float s = 0.0f, s2 = 0.0f;
    for (int c = tid; c < CC; c += 256) {
        float v = in[row * CC + c];
        s += v; s2 += v * v;
    }
#pragma unroll
    for (int off = 16; off > 0; off >>= 1) {
        s  += __shfl_xor(s, off, 32);
        s2 += __shfl_xor(s2, off, 32);
    }
    __shared__ float ss[8], ss2[8];
    const int wid = tid >> 5, lane = tid & 31;
    if (lane == 0) { ss[wid] = s; ss2[wid] = s2; }
    __syncthreads();
    if (tid == 0) {
        float a = 0.0f, b2 = 0.0f;
#pragma unroll
        for (int i = 0; i < 8; ++i) { a += ss[i]; b2 += ss2[i]; }
        ss[0] = a; ss2[0] = b2;
    }
    __syncthreads();
    const float mean = ss[0] * (1.0f / CC);
    const float var  = ss2[0] * (1.0f / CC) - mean * mean;
    const float inv  = rsqrtf(var + 1e-5f);
    for (int c = tid; c < CC; c += 256) {
        float v = in[row * CC + c];
        out[row * CC + c] = (v - mean) * inv * w[c] + bb[c];
    }
}

// ---------------- host driver ----------------
extern "C" void kernel_launch(void* const* d_in, const int* in_sizes, int n_in,
                              void* d_out, int out_size, void* d_ws, size_t ws_size,
                              hipStream_t stream) {
    const float* x      = (const float*)d_in[0];
    const float* wq     = (const float*)d_in[1];
    const float* wk     = (const float*)d_in[2];
    const float* wv     = (const float*)d_in[3];
    const float* wo     = (const float*)d_in[4];
    const float* conv_w = (const float*)d_in[5];
    const float* conv_b = (const float*)d_in[6];
    const float* gate_w = (const float*)d_in[7];
    const float* gate_b = (const float*)d_in[8];
    const float* ln_w   = (const float*)d_in[9];
    const float* ln_b   = (const float*)d_in[10];

    float* out_y     = (float*)d_out;                        // [B,T,C]
    float* out_state = (float*)d_out + (long)BB * TT * CC;   // [B,H,D]

    char* ws = (char*)d_ws;
    size_t off = 0;
    auto carve = [&](size_t bytes) -> char* {
        char* p = ws + off;
        off = (off + bytes + 255) & ~(size_t)255;
        return p;
    };
    const long NTC = (long)BB * TT * CC;
    const long WSZ = (long)CC * CC;
    __bf16* wqbf = (__bf16*)carve(WSZ * 2);
    __bf16* wkbf = (__bf16*)carve(WSZ * 2);
    __bf16* wvbf = (__bf16*)carve(WSZ * 2);
    __bf16* wobf = (__bf16*)carve(WSZ * 2);
    __bf16* gwbf = (__bf16*)carve(WSZ * 2);
    __bf16* xbf  = (__bf16*)carve(NTC * 2);
    __bf16* xcbf = (__bf16*)carve(NTC * 2);
    float*  qb   = (float*)carve(NTC * 4);
    float*  kb   = (float*)carve(NTC * 4);
    float*  vb   = (float*)carve(NTC * 4);   // becomes kv in-place
    float*  gb   = (float*)carve(NTC * 4);   // gate, reused as GEMM-out buffer
    __bf16* ubf  = (__bf16*)carve(NTC * 2);
    (void)ws_size;

    // 1) weights -> bf16
    {
        dim3 g((unsigned)((WSZ + 255) / 256)), blk(256);
        cvt_bf16_kernel<<<g, blk, 0, stream>>>(wq, wqbf, WSZ);
        cvt_bf16_kernel<<<g, blk, 0, stream>>>(wk, wkbf, WSZ);
        cvt_bf16_kernel<<<g, blk, 0, stream>>>(wv, wvbf, WSZ);
        cvt_bf16_kernel<<<g, blk, 0, stream>>>(wo, wobf, WSZ);
        cvt_bf16_kernel<<<g, blk, 0, stream>>>(gate_w, gwbf, WSZ);
    }
    // 2) causal dwconv + x -> bf16
    conv_kernel<<<(unsigned)((NTC + 255) / 256), 256, 0, stream>>>(
        x, conv_w, conv_b, xbf, xcbf);

    // 3) projections (TDM-staged WMMA bf16, f32 accumulate)
    dim3 ggrid(CC / 128, MM / 128);  // (16, 64)
    gemm_bf16_nt<<<ggrid, 256, 0, stream>>>(xbf,  wqbf, qb, nullptr, MM, CC, CC, 0);
    gemm_bf16_nt<<<ggrid, 256, 0, stream>>>(xbf,  wkbf, kb, nullptr, MM, CC, CC, 0);
    gemm_bf16_nt<<<ggrid, 256, 0, stream>>>(xbf,  wvbf, vb, nullptr, MM, CC, CC, 0);
    gemm_bf16_nt<<<ggrid, 256, 0, stream>>>(xcbf, gwbf, gb, gate_b,  MM, CC, CC, 1);

    // 4) l2norm(q), l2norm(k), kv = gate*kn*v
    {
        long rows = (long)BB * TT * HH;       // 131072, 8 rows / block
        qknorm_kv_kernel<<<(unsigned)(rows / 8), 256, 0, stream>>>(qb, kb, vb, gb);
    }
    // 5) EMA scan fused with u = mem*q -> bf16, next_state -> d_out tail
    scan_kernel<<<(BB * CC) / 256, 256, 0, stream>>>(vb, qb, ubf, out_state);

    // 6) output projection (reuse gb as f32 output buffer)
    gemm_bf16_nt<<<ggrid, 256, 0, stream>>>(ubf, wobf, gb, nullptr, MM, CC, CC, 0);

    // 7) LayerNorm -> y
    layernorm_kernel<<<(unsigned)((long)BB * TT), 256, 0, stream>>>(gb, ln_w, ln_b, out_y);
}
